// DTFN_84653805404444
// MI455X (gfx1250) — matrix operand order repaired and checked by reading, not compiled, not run on hardware
//
#include <hip/hip_runtime.h>
#include <hip/hip_bf16.h>

// Problem constants (match reference)
#define EPSV   1e-6f
#define DD     512
#define SS     2048
#define VV     32000
#define MROWS  4096   // B*S
#define MFLUX  4094   // B*(S-1)
#define SM1    2047   // S-1
#define TK     128    // K-chunk staged in LDS for the decoder GEMM

// Async global->LDS path (ASYNCcnt) if this toolchain exposes the builtins;
// otherwise fall back to global->VGPR->ds_store staging (known-good).
#if defined(__has_builtin)
#  if __has_builtin(__builtin_amdgcn_global_load_async_to_lds_b128)
#    define ASYNC_LDS 1
#  endif
#endif
#ifndef ASYNC_LDS
#  define ASYNC_LDS 0
#endif

typedef __attribute__((ext_vector_type(16))) _Float16 v16h;
typedef __attribute__((ext_vector_type(8)))  _Float16 v8h;
typedef __attribute__((ext_vector_type(8)))  float    v8f;
typedef __attribute__((ext_vector_type(4)))  int      v4i;

// Address-space-qualified b128 pointers for the async-to-LDS builtin:
// arg0 is AS1 (global, "__device__") int4*, arg1 is AS3 (LDS) pointer.
typedef __attribute__((address_space(1))) v4i gas_v4i;
typedef __attribute__((address_space(3))) v4i las_v4i;

__device__ __forceinline__ float softplus_f(float x) {
    // numerically stable softplus = logaddexp(x, 0)
    return x > 0.f ? x + log1pf(expf(-x)) : log1pf(expf(x));
}

// ---------------------------------------------------------------------------
// A-fragment: 16x32 f16 tile, contiguous k order matching the B loads (wave32).
// lane L owns row M = L%16; g = L/16:
//   halves j=0..7  -> K = kb + g*16 + j
//   halves j=8..15 -> K = kb + g*16 + 8 + (j-8)
// ---------------------------------------------------------------------------
__device__ __forceinline__ v16h load_a_frag(const _Float16* rowp, int kb, int g) {
    v8h lo = *(const v8h*)(rowp + kb + g * 16);
    v8h hi = *(const v8h*)(rowp + kb + g * 16 + 8);
    v16h a;
#pragma unroll
    for (int i = 0; i < 8; ++i) { a[i] = lo[i]; a[i + 8] = hi[i]; }
    return a;
}

// ---------------------------------------------------------------------------
// Tiled LDS transpose + f32->f16 convert: w [K x N] row-major -> wT [N x K].
// ---------------------------------------------------------------------------
__global__ void transpose_to_f16(const float* __restrict__ w,
                                 _Float16* __restrict__ wT, int K, int N) {
    __shared__ float tile[32][33];
    const int bk = blockIdx.x * 32, bn = blockIdx.y * 32;
    const int tx = threadIdx.x, ty = threadIdx.y;  // block (32, 8)
#pragma unroll
    for (int i = ty; i < 32; i += 8)
        tile[i][tx] = w[(size_t)(bk + i) * N + (bn + tx)];
    __syncthreads();
#pragma unroll
    for (int i = ty; i < 32; i += 8)
        wT[(size_t)(bn + i) * K + (bk + tx)] = (_Float16)tile[tx][i];
}

// Embedding gather + f16 convert: x16[r, d] = (f16) emb[x[r], d]
__global__ void embed_gather(const int* __restrict__ x,
                             const float* __restrict__ emb,
                             _Float16* __restrict__ x16) {
    const int i = blockIdx.x * blockDim.x + threadIdx.x;  // over MROWS*DD
    const int r = i >> 9, d = i & (DD - 1);
    x16[i] = (_Float16)emb[(size_t)x[r] * DD + d];
}

// m = clip(m + dt*(F[s-1] - F[s]), EPS)   with zero-flux boundaries
__global__ void flux_update(float* __restrict__ m32,
                            _Float16* __restrict__ m16,
                            const float* __restrict__ Fbuf,
                            const float* __restrict__ cfl) {
    const int i = blockIdx.x * blockDim.x + threadIdx.x;  // over MROWS*DD
    const int row = i >> 9, d = i & (DD - 1);
    const int b = row >> 11, s = row & (SS - 1);
    const float dt = 1.f / (1.f + expf(-cfl[0]));
    const float Fl = (s > 0)      ? Fbuf[((size_t)(b * SM1 + s - 1) << 9) + d] : 0.f;
    const float Fr = (s < SS - 1) ? Fbuf[((size_t)(b * SM1 + s) << 9) + d]     : 0.f;
    float v = m32[i] + dt * (Fl - Fr);
    v = v < EPSV ? EPSV : v;
    m32[i] = v;
    m16[i] = (_Float16)v;
}

// ---------------------------------------------------------------------------
// D=512 GEMMs (mass / flux MLP): wave -> 16M x 64N, direct-from-global.
// ---------------------------------------------------------------------------
enum { MODE_MASS = 0, MODE_F1 = 1, MODE_F2 = 2 };

template <int MODE>
__global__ __launch_bounds__(128)
void gemm_wmma(const _Float16* __restrict__ A,
               const _Float16* __restrict__ WT,   // [N x K] f16 (transposed)
               const float* __restrict__ bias,    // [N]
               float* __restrict__ outF,
               _Float16* __restrict__ outH,
               int K, int N) {
    const int lane  = threadIdx.x;
    const int n15   = lane & 15;
    const int g     = lane >> 4;
    const int mtile = blockIdx.x;
    const int nbase = blockIdx.y * 256 + threadIdx.y * 64;

    // Per-lane A row pointer(s)
    const int rowA = mtile * 16 + n15;
    const _Float16* pA0;
    const _Float16* pA1 = nullptr;
    if (MODE == MODE_F1) {
        // flux row rf -> m row b*S + s ; second K-half uses next cell (s+1)
        int rf = rowA > MFLUX - 1 ? MFLUX - 1 : rowA;
        int b = rf / SM1, s = rf - b * SM1;
        pA0 = A + (size_t)(b * SS + s) * DD;
        pA1 = pA0 + DD;
    } else {
        pA0 = A + (size_t)rowA * K;
    }

    const _Float16* colp[4];
#pragma unroll
    for (int t = 0; t < 4; ++t)
        colp[t] = WT + (size_t)(nbase + t * 16 + n15) * (size_t)K + (size_t)(g * 16);

    v8f acc[4] = {v8f{}, v8f{}, v8f{}, v8f{}};

    for (int kb = 0; kb < K; kb += 32) {
        v16h a;
        if (MODE == MODE_F1) {
            const _Float16* rp = (kb < DD) ? pA0 : pA1;
            a = load_a_frag(rp, kb & (DD - 1), g);
        } else {
            a = load_a_frag(pA0, kb, g);
        }
#pragma unroll
        for (int t = 0; t < 4; ++t) {
            v16h bfr = *(const v16h*)(colp[t] + kb);
            acc[t] = __builtin_amdgcn_wmma_f32_16x16x32_f16(
                false, a, false, bfr, (short)0, acc[t], false, false);
        }
    }

#pragma unroll
    for (int t = 0; t < 4; ++t) {
        const int n = nbase + t * 16 + n15;
        const float bv = bias[n];
#pragma unroll
        for (int r = 0; r < 8; ++r) {
            const int row = mtile * 16 + r + 8 * g;
            float v = acc[t][r] + bv;
            if (MODE == MODE_MASS) {
                v = softplus_f(v) + EPSV;
                outF[(size_t)row * N + n] = v;
                outH[(size_t)row * N + n] = (_Float16)v;
            } else if (MODE == MODE_F1) {
                outH[(size_t)row * N + n] = (_Float16)tanhf(v);
            } else {  // MODE_F2
                outF[(size_t)row * N + n] = softplus_f(v);
            }
        }
    }
}

// ---------------------------------------------------------------------------
// Decoder GEMM (4096 x 512 x 32000): block = 256 threads = 8 waves covering
// 128M x 128N. A and B staged in LDS per 128-wide K-chunk (async global->LDS
// when available); each wave computes 32M x 64N (2 A-frags x 4 B-frags ->
// 8 WMMA per 32-K step) so staged bytes are reused 8x.
// Logits leave via non-temporal stores (write-once stream; keep L2 for B).
// ---------------------------------------------------------------------------
__global__ __launch_bounds__(256)
void gemm_dec_lds(const _Float16* __restrict__ A,    // m16 [4096 x 512]
                  const _Float16* __restrict__ WT,   // wdecT [32000 x 512]
                  const float* __restrict__ bias,    // [32000]
                  float* __restrict__ out) {         // [4096 x 32000]
    __shared__ alignas(32) _Float16 sA[128 * TK];    // row-major [m][k], 32 KB
    __shared__ alignas(32) _Float16 sB[128 * TK];    // col-major [n][k], 32 KB

    const int tid  = threadIdx.x;       // 0..255
    const int lane = tid & 31;
    const int wv   = tid >> 5;          // 0..7
    const int n15  = lane & 15;
    const int g    = lane >> 4;
    const int mblk = blockIdx.x * 128;
    const int nblk = blockIdx.y * 128;
    const int mw   = (wv & 3) * 32;     // wave's M offset inside tile
    const int nw   = (wv >> 2) * 64;    // wave's N offset inside tile

    // Staging: thread t copies 64 contiguous halves (128 B) of one row/col.
    const int srow = tid >> 1;
    const int skof = (tid & 1) * 64;
    const _Float16* gA = A  + (size_t)(mblk + srow) * DD + skof;
    const _Float16* gB = WT + (size_t)(nblk + srow) * DD + skof;
    _Float16* lA = sA + srow * TK + skof;
    _Float16* lB = sB + srow * TK + skof;

    v8f acc[2][4];
#pragma unroll
    for (int i = 0; i < 2; ++i)
#pragma unroll
        for (int t = 0; t < 4; ++t) acc[i][t] = v8f{};

    for (int kb = 0; kb < DD; kb += TK) {
        __syncthreads();  // previous chunk's LDS reads complete
#if ASYNC_LDS
        // 8 x 16B async transfers each for A and B: cache -> LDS, no VGPRs,
        // tracked by ASYNCcnt.
#pragma unroll
        for (int c = 0; c < 8; ++c) {
            __builtin_amdgcn_global_load_async_to_lds_b128(
                (gas_v4i*)(gA + kb + c * 8), (las_v4i*)(lA + c * 8), 0, 0);
            __builtin_amdgcn_global_load_async_to_lds_b128(
                (gas_v4i*)(gB + kb + c * 8), (las_v4i*)(lB + c * 8), 0, 0);
        }
#  if defined(__has_builtin) && __has_builtin(__builtin_amdgcn_s_wait_asynccnt)
        __builtin_amdgcn_s_wait_asynccnt(0);
#  else
        asm volatile("s_wait_asynccnt 0" ::: "memory");
#  endif
#else
#pragma unroll
        for (int c = 0; c < 4; ++c) {   // 4 x 32B vector copies each for A, B
            *(v16h*)(lA + c * 16) = *(const v16h*)(gA + kb + c * 16);
            *(v16h*)(lB + c * 16) = *(const v16h*)(gB + kb + c * 16);
        }
#endif
        if (kb + TK < DD) {             // prefetch next chunk into cache
            __builtin_prefetch(gA + kb + TK, 0, 3);
            __builtin_prefetch(gB + kb + TK, 0, 3);
        }
        __syncthreads();

#pragma unroll
        for (int kk = 0; kk < TK; kk += 32) {
            v16h afr[2], bfr[4];
#pragma unroll
            for (int i = 0; i < 2; ++i)
                afr[i] = load_a_frag(sA + (mw + i * 16 + n15) * TK, kk, g);
#pragma unroll
            for (int t = 0; t < 4; ++t)
                bfr[t] = *(const v16h*)(sB + (nw + t * 16 + n15) * TK + kk + g * 16);
#pragma unroll
            for (int i = 0; i < 2; ++i)
#pragma unroll
                for (int t = 0; t < 4; ++t)
                    acc[i][t] = __builtin_amdgcn_wmma_f32_16x16x32_f16(
                        false, afr[i], false, bfr[t], (short)0, acc[i][t],
                        false, false);
        }
    }

    // Epilogue: lane owns N = n15; VGPR r -> M = r + 8*g. Non-temporal stores.
#pragma unroll
    for (int t = 0; t < 4; ++t) {
        const int n = nblk + nw + t * 16 + n15;
        const float bv = bias[n];
#pragma unroll
        for (int i = 0; i < 2; ++i) {
#pragma unroll
            for (int r = 0; r < 8; ++r) {
                const int row = mblk + mw + i * 16 + r + 8 * g;
                float v = acc[i][t][r] + bv;
                __builtin_nontemporal_store(v, &out[(size_t)row * VV + n]);
            }
        }
    }
}

extern "C" void kernel_launch(void* const* d_in, const int* in_sizes, int n_in,
                              void* d_out, int out_size, void* d_ws, size_t ws_size,
                              hipStream_t stream) {
    (void)in_sizes; (void)n_in; (void)out_size; (void)ws_size;

    const int*   x     = (const int*)  d_in[0];
    const float* emb   = (const float*)d_in[1];
    const float* wmass = (const float*)d_in[2];
    const float* bmass = (const float*)d_in[3];
    const float* wf1   = (const float*)d_in[4];
    const float* bf1   = (const float*)d_in[5];
    const float* wf2   = (const float*)d_in[6];
    const float* bf2   = (const float*)d_in[7];
    const float* cfl   = (const float*)d_in[8];
    const float* wdec  = (const float*)d_in[9];
    const float* bdec  = (const float*)d_in[10];
    float* out = (float*)d_out;

    // Workspace carve-up (~63 MB total)
    char* p = (char*)d_ws;
    auto take = [&](size_t bytes) { void* q = p; p += (bytes + 255) & ~(size_t)255; return q; };
    float*    m32  = (float*)   take((size_t)MROWS * DD * 4);
    _Float16* m16  = (_Float16*)take((size_t)MROWS * DD * 2);
    _Float16* x16  = (_Float16*)take((size_t)MROWS * DD * 2);
    _Float16* hid  = (_Float16*)take((size_t)MROWS * DD * 2);   // padded flux rows
    float*    Fbuf = (float*)   take((size_t)MROWS * DD * 4);   // padded flux rows
    _Float16* wmT  = (_Float16*)take((size_t)DD * DD * 2);
    _Float16* wf1T = (_Float16*)take((size_t)DD * 2 * DD * 2);
    _Float16* wf2T = (_Float16*)take((size_t)DD * DD * 2);
    _Float16* wdT  = (_Float16*)take((size_t)VV * DD * 2);

    const dim3 tb(32, 8);
    transpose_to_f16<<<dim3(DD / 32,     DD / 32), tb, 0, stream>>>(wmass, wmT,  DD,     DD);
    transpose_to_f16<<<dim3(2 * DD / 32, DD / 32), tb, 0, stream>>>(wf1,   wf1T, 2 * DD, DD);
    transpose_to_f16<<<dim3(DD / 32,     DD / 32), tb, 0, stream>>>(wf2,   wf2T, DD,     DD);
    transpose_to_f16<<<dim3(DD / 32,     VV / 32), tb, 0, stream>>>(wdec,  wdT,  DD,     VV);

    embed_gather<<<(MROWS * DD) / 256, 256, 0, stream>>>(x, emb, x16);

    const dim3 gb(32, 4);
    // m0 = softplus(emb[x] @ w_mass + b_mass) + EPS
    gemm_wmma<MODE_MASS><<<dim3(MROWS / 16, DD / 256), gb, 0, stream>>>(
        x16, wmT, bmass, m32, m16, DD, DD);

    for (int it = 0; it < 3; ++it) {
        // hid = tanh(concat(m[:-1], m[1:]) @ w_f1 + b_f1)
        gemm_wmma<MODE_F1><<<dim3(MROWS / 16, DD / 256), gb, 0, stream>>>(
            m16, wf1T, bf1, nullptr, hid, 2 * DD, DD);
        // F = softplus(hid @ w_f2 + b_f2)
        gemm_wmma<MODE_F2><<<dim3(MROWS / 16, DD / 256), gb, 0, stream>>>(
            hid, wf2T, bf2, Fbuf, nullptr, DD, DD);
        // m = clip(m + dt*(F[s-1]-F[s]), EPS)
        flux_update<<<(MROWS * DD) / 256, 256, 0, stream>>>(m32, m16, Fbuf, cfl);
    }

    // logits = m @ w_dec + b_dec  (LDS-tiled, async staging, NT stores)
    gemm_dec_lds<<<dim3(MROWS / 128, VV / 128), 256, 0, stream>>>(m16, wdT, bdec, out);
}
